// pendulumRNN3_15307263443606
// MI455X (gfx1250) — matrix operand order repaired and checked by reading, not compiled, not run on hardware
//
#include <hip/hip_runtime.h>

#define Hdim 256
#define Tlen 512
#define Bsz  256
#define G4H  1024                 // 4*H
#define NWELEM (G4H * Hdim)       // 262144 elements per weight matrix

typedef __attribute__((ext_vector_type(16))) __bf16 v16bf;
typedef __attribute__((ext_vector_type(8)))  __bf16 v8bf;
typedef __attribute__((ext_vector_type(8)))  float  v8f;

__device__ __forceinline__ unsigned short f2bf(float f) {
    union { float f; unsigned u; } v; v.f = f;
    unsigned r = v.u + 0x7FFFu + ((v.u >> 16) & 1u);   // round-to-nearest-even
    return (unsigned short)(r >> 16);
}
__device__ __forceinline__ float bf2f(unsigned short b) {
    union { unsigned u; float f; } v; v.u = ((unsigned)b) << 16;
    return v.f;
}
// fast activations: v_exp_f32 + v_rcp_f32, no IEEE divide expansion
__device__ __forceinline__ float sigm(float x) {
    return __builtin_amdgcn_rcpf(1.0f + __expf(-x));
}
__device__ __forceinline__ float tanh_f(float x) {
    float e = __expf(-2.0f * fabsf(x));
    float t = (1.0f - e) * __builtin_amdgcn_rcpf(1.0f + e);
    return x >= 0.0f ? t : -t;
}

// ---------------------------------------------------------------------------
// Prep: fp32 weights -> bf16 in workspace (row-major kept: B-fragment of
// gates = h @ W^T wants K runs [lh*8,+8) and +16 of W row n per lane).
// Also fold biases: bias = b_ih + b_hh, and copy the rank-1 layer-1 input col.
// ---------------------------------------------------------------------------
__global__ void prep_kernel(const float* __restrict__ Wih1, const float* __restrict__ Whh1,
                            const float* __restrict__ bih1, const float* __restrict__ bhh1,
                            const float* __restrict__ Wih2, const float* __restrict__ Whh2,
                            const float* __restrict__ bih2, const float* __restrict__ bhh2,
                            const float* __restrict__ Wih3, const float* __restrict__ Whh3,
                            const float* __restrict__ bih3, const float* __restrict__ bhh3,
                            unsigned short* __restrict__ wbf,
                            float* __restrict__ bias1, float* __restrict__ wi1,
                            float* __restrict__ bias2, float* __restrict__ bias3) {
    int idx = blockIdx.x * blockDim.x + threadIdx.x;
    if (idx < NWELEM)            wbf[idx] = f2bf(Whh1[idx]);
    else if (idx < 2 * NWELEM)   wbf[idx] = f2bf(Wih2[idx - 1 * NWELEM]);
    else if (idx < 3 * NWELEM)   wbf[idx] = f2bf(Whh2[idx - 2 * NWELEM]);
    else if (idx < 4 * NWELEM)   wbf[idx] = f2bf(Wih3[idx - 3 * NWELEM]);
    else if (idx < 5 * NWELEM)   wbf[idx] = f2bf(Whh3[idx - 4 * NWELEM]);
    if (idx < G4H) {
        bias1[idx] = bih1[idx] + bhh1[idx];
        wi1[idx]   = Wih1[idx];
        bias2[idx] = bih2[idx] + bhh2[idx];
        bias3[idx] = bih3[idx] + bhh3[idx];
    }
}

// ---------------------------------------------------------------------------
// One K=256 WMMA pass: acc[g] += A(ldsA 16xH bf16) x W^T tile (gate g, col w).
// A frag (16x32 bf16): lane half lh -> K chunks [lh*8, +8) and +16.
// B frag (32x16 bf16): lane -> column n = lane%16; bf16 of W row n
//   (row-major), K chunks [lh*8, +8) and +16 (same as A).
// kt loop kept rolled (#pragma unroll 1) so only one k-step of B fragments
// is live at a time -> no spills.
// ---------------------------------------------------------------------------
__device__ __forceinline__ void mma_pass(v8f (&acc)[4],
                                         const unsigned short* ldsA,
                                         const unsigned short* __restrict__ W,
                                         int lane, int w) {
    const int lh = lane >> 4;
    const int ln = lane & 15;
    const unsigned short* Arow = ldsA + ln * Hdim + lh * 8;
#pragma unroll 1
    for (int kt = 0; kt < 8; ++kt) {
        const int k0 = kt * 32;
        union { v16bf v; v8bf h[2]; } a;
        a.h[0] = *(const v8bf*)(Arow + k0);
        a.h[1] = *(const v8bf*)(Arow + k0 + 16);
#pragma unroll
        for (int g = 0; g < 4; ++g) {
            const int n = (g * 16 + w) * 16 + ln;       // W row (gate output idx)
            union { v16bf v; v8bf h[2]; } b;
            b.h[0] = *(const v8bf*)(W + n * Hdim + k0 + lh * 8);
            b.h[1] = *(const v8bf*)(W + n * Hdim + k0 + lh * 8 + 16);
            acc[g] = __builtin_amdgcn_wmma_f32_16x16x32_bf16(
                false, a.v, false, b.v, (short)0, acc[g], false, false);
        }
    }
}

// Gate nonlinearities + cell update; writes h_new (bf16) into LDS.
// C/D layout: VGPR r -> (M = r + lh*8, N = w*16 + lane%16).
__device__ __forceinline__ void lstm_elem(v8f (&acc)[4], float (&cst)[8],
                                          unsigned short* hout, int lane, int w) {
    const int lh   = lane >> 4;
    const int ln   = lane & 15;
    const int ncol = w * 16 + ln;                       // column within H
#pragma unroll
    for (int r = 0; r < 8; ++r) {
        float iv = sigm(acc[0][r]);
        float fv = sigm(acc[1][r]);
        float gv = tanh_f(acc[2][r]);
        float ov = sigm(acc[3][r]);
        float c  = fv * cst[r] + iv * gv;
        cst[r]   = c;
        float h  = ov * tanh_f(c);
        const int m = r + lh * 8;
        hout[m * Hdim + ncol] = f2bf(h);
    }
}

// ---------------------------------------------------------------------------
// Persistent LSTM: one workgroup per 16-row batch tile, loops all T steps.
// 16 waves (512 threads); wave w owns gate column-tile w of each i/f/g/o
// region (4 accumulators), so the cell state never leaves registers.
// ---------------------------------------------------------------------------
__global__ __launch_bounds__(512, 1)
void lstm_kernel(const float* __restrict__ input,
                 const unsigned short* __restrict__ Wh1,
                 const unsigned short* __restrict__ Wi2,
                 const unsigned short* __restrict__ Wh2,
                 const unsigned short* __restrict__ Wi3,
                 const unsigned short* __restrict__ Wh3,
                 const float* __restrict__ bias1, const float* __restrict__ wi1,
                 const float* __restrict__ bias2, const float* __restrict__ bias3,
                 const float* __restrict__ Wlin, const float* __restrict__ blin,
                 float* __restrict__ out) {
    __shared__ __align__(16) unsigned short h1[16 * Hdim];
    __shared__ __align__(16) unsigned short h2[16 * Hdim];
    __shared__ __align__(16) unsigned short h3[16 * Hdim];
    __shared__ float xbuf[16];
    __shared__ float ypart[16][32];

    const int tid  = threadIdx.x;
    const int lane = tid & 31;
    const int w    = tid >> 5;          // wave id 0..15 = gate column tile
    const int lh   = lane >> 4;
    const int ln   = lane & 15;
    const int b0   = blockIdx.x * 16;

    for (int i = tid; i < 16 * Hdim; i += 512) { h1[i] = 0; h2[i] = 0; h3[i] = 0; }

    // loop-invariant per-lane constants (biases + rank-1 layer-1 input column)
    float b1r[4], w1r[4], b2r[4], b3r[4];
#pragma unroll
    for (int g = 0; g < 4; ++g) {
        const int n = (g * 16 + w) * 16 + ln;
        b1r[g] = bias1[n];
        w1r[g] = wi1[n];
        b2r[g] = bias2[n];
        b3r[g] = bias3[n];
    }
    // output head mapping: 16 rows x 32 segments of 8
    const int yrow = tid >> 5;
    const int yseg = tid & 31;
    float wl[8];
#pragma unroll
    for (int i = 0; i < 8; ++i) wl[i] = Wlin[yseg * 8 + i];
    const float bl = blin[0];

    float c1[8], c2[8], c3[8];
#pragma unroll
    for (int r = 0; r < 8; ++r) { c1[r] = 0.f; c2[r] = 0.f; c3[r] = 0.f; }

    __syncthreads();

    for (int t = 0; t < Tlen; ++t) {
        if (tid < 16) xbuf[tid] = input[(b0 + tid) * Tlen + t];
        __syncthreads();
        float xr[8];
#pragma unroll
        for (int r = 0; r < 8; ++r) xr[r] = xbuf[r + lh * 8];

        v8f acc[4];

        // ---- layer 1: C preload = bias + x * W_ih1 column (input_size == 1)
#pragma unroll
        for (int g = 0; g < 4; ++g)
#pragma unroll
            for (int r = 0; r < 8; ++r)
                acc[g][r] = b1r[g] + xr[r] * w1r[g];
        mma_pass(acc, h1, Wh1, lane, w);
        __syncthreads();                 // all waves done reading h1
        lstm_elem(acc, c1, h1, lane, w); // write h1_new
        __syncthreads();

        // ---- layer 2
#pragma unroll
        for (int g = 0; g < 4; ++g)
#pragma unroll
            for (int r = 0; r < 8; ++r)
                acc[g][r] = b2r[g];
        mma_pass(acc, h1, Wi2, lane, w);
        mma_pass(acc, h2, Wh2, lane, w);
        __syncthreads();
        lstm_elem(acc, c2, h2, lane, w);
        __syncthreads();

        // ---- layer 3
#pragma unroll
        for (int g = 0; g < 4; ++g)
#pragma unroll
            for (int r = 0; r < 8; ++r)
                acc[g][r] = b3r[g];
        mma_pass(acc, h2, Wi3, lane, w);
        mma_pass(acc, h3, Wh3, lane, w);
        __syncthreads();
        lstm_elem(acc, c3, h3, lane, w);
        __syncthreads();

        // ---- output head: y = h3 @ W_lin^T + b_lin  (16 rows x 32 segs of 8)
        {
            float s = 0.f;
            const unsigned short* hp = h3 + yrow * Hdim + yseg * 8;
#pragma unroll
            for (int i = 0; i < 8; ++i) s += bf2f(hp[i]) * wl[i];
            ypart[yrow][yseg] = s;
            __syncthreads();
            if (yseg == 0) {
                float tot = bl;
#pragma unroll
                for (int i = 0; i < 32; ++i) tot += ypart[yrow][i];
                out[(b0 + yrow) * Tlen + t] = tot;
            }
            __syncthreads();
        }
    }
}

extern "C" void kernel_launch(void* const* d_in, const int* in_sizes, int n_in,
                              void* d_out, int out_size, void* d_ws, size_t ws_size,
                              hipStream_t stream) {
    (void)in_sizes; (void)n_in; (void)out_size; (void)ws_size;
    const float* input = (const float*)d_in[0];
    const float* Wih1  = (const float*)d_in[1];
    const float* Whh1  = (const float*)d_in[2];
    const float* bih1  = (const float*)d_in[3];
    const float* bhh1  = (const float*)d_in[4];
    const float* Wih2  = (const float*)d_in[5];
    const float* Whh2  = (const float*)d_in[6];
    const float* bih2  = (const float*)d_in[7];
    const float* bhh2  = (const float*)d_in[8];
    const float* Wih3  = (const float*)d_in[9];
    const float* Whh3  = (const float*)d_in[10];
    const float* bih3  = (const float*)d_in[11];
    const float* bhh3  = (const float*)d_in[12];
    const float* Wlin  = (const float*)d_in[13];
    const float* blin  = (const float*)d_in[14];
    float* out = (float*)d_out;

    // workspace: 5 bf16 weight matrices (2.62 MB) + 4 fp32 bias/col arrays (16 KB)
    unsigned short* wbf = (unsigned short*)d_ws;
    float* bias1 = (float*)(wbf + 5 * NWELEM);
    float* wi1   = bias1 + G4H;
    float* bias2 = wi1 + G4H;
    float* bias3 = bias2 + G4H;

    prep_kernel<<<(5 * NWELEM + 255) / 256, 256, 0, stream>>>(
        Wih1, Whh1, bih1, bhh1, Wih2, Whh2, bih2, bhh2,
        Wih3, Whh3, bih3, bhh3, wbf, bias1, wi1, bias2, bias3);

    lstm_kernel<<<Bsz / 16, 512, 0, stream>>>(
        input, wbf, wbf + 1 * NWELEM, wbf + 2 * NWELEM, wbf + 3 * NWELEM, wbf + 4 * NWELEM,
        bias1, wi1, bias2, bias3, Wlin, blin, out);
}
